// Quantize_90821378441404
// MI455X (gfx1250) — matrix operand order repaired and checked by reading, not compiled, not run on hardware
//
#include <hip/hip_runtime.h>
#include <hip/hip_bf16.h>

typedef __attribute__((ext_vector_type(16))) __bf16 v16bf;
typedef __attribute__((ext_vector_type(8)))  float  v8f;
typedef unsigned int u32x4 __attribute__((ext_vector_type(4)));
typedef int          i32x8 __attribute__((ext_vector_type(8)));
typedef int          i32x4 __attribute__((ext_vector_type(4)));

#define N_TOK     16384
#define DIM_D     256
#define DIM_K     8192
#define DIFF_OFF  4194304
#define IND_OFF   4194305

// ---------------------------------------------------------------------------
// Kernel 1: transpose + convert embed [D=256][K=8192] f32 -> eb_t [K][D] bf16
// ---------------------------------------------------------------------------
__global__ __launch_bounds__(256) void vq_transpose_convert(
    const float* __restrict__ embed, __bf16* __restrict__ eb_t) {
  __shared__ float tile[32][33];
  const int k0 = blockIdx.x * 32;
  const int d0 = blockIdx.y * 32;
  const int tx = threadIdx.x;   // 0..31
  const int ty = threadIdx.y;   // 0..7
#pragma unroll
  for (int i = 0; i < 4; ++i) {
    int d = d0 + ty + i * 8;
    tile[ty + i * 8][tx] = embed[(size_t)d * DIM_K + k0 + tx];
  }
  __syncthreads();
#pragma unroll
  for (int i = 0; i < 4; ++i) {
    int k = k0 + ty + i * 8;
    eb_t[(size_t)k * DIM_D + d0 + tx] = (__bf16)tile[tx][ty + i * 8];
  }
}

// ---------------------------------------------------------------------------
// Kernel 2: enorm[k] = sum_d embed[d][k]^2   (exact f32, matches reference)
// ---------------------------------------------------------------------------
__global__ __launch_bounds__(256) void vq_enorm(
    const float* __restrict__ embed, float* __restrict__ enorm) {
  const int k = blockIdx.x * blockDim.x + threadIdx.x;
  float s = 0.f;
  for (int d = 0; d < DIM_D; ++d) {
    float v = embed[(size_t)d * DIM_K + k];
    s = fmaf(v, v, s);
  }
  enorm[k] = s;
}

// ---------------------------------------------------------------------------
// TDM: 1-D contiguous 16 KB copy, global -> LDS.
// D# per CDNA5 ISA ch.8: data_size=3 (8B elems), tile_dim0 = tensor_dim0 =
// tensor_dim0_stride = 2048, 1 row. Groups 2/3 unused (zero).
// 6-arg builtin form (clang-23 / therock-10.0 headers).
// ---------------------------------------------------------------------------
__device__ __forceinline__ void tdm_load_16k(const void* gsrc,
                                             unsigned ldsOff) {
  unsigned long long ga = (unsigned long long)(uintptr_t)gsrc;
  u32x4 g0;
  g0[0] = 1u;                                   // count=1, user descriptor
  g0[1] = ldsOff;                               // lds_addr (bytes)
  g0[2] = (unsigned)(ga & 0xFFFFFFFFull);       // global_addr[31:0]
  g0[3] = (unsigned)((ga >> 32) & 0x1FFFFFFull) // global_addr[56:32]
          | (2u << 30);                         // type=2 ("image")
  i32x8 g1;
  g1[0] = 0x00030000;  // workgroup_mask=0, data_size=3 (8 bytes)
  g1[1] = 0x08000000;  // tensor_dim0 = 2048 (bits 79:48, low half)
  g1[2] = 0x00010000;  // tensor_dim0 hi = 0, tensor_dim1 = 1 (low half)
  g1[3] = 0x08000000;  // tensor_dim1 hi = 0, tile_dim0 = 2048
  g1[4] = 0;           // tile_dim1 = 0 (unused), tile_dim2 = 0
  g1[5] = 0x00000800;  // tensor_dim0_stride = 2048 (low 32)
  g1[6] = 0;           // stride hi, tensor_dim1_stride lo
  g1[7] = 0;
  i32x4 z4 = {0, 0, 0, 0};
  i32x8 z8 = {0, 0, 0, 0, 0, 0, 0, 0};
  __builtin_amdgcn_tensor_load_to_lds(g0, g1, z4, z4, z8, 0);
}

// ---------------------------------------------------------------------------
// Kernel 3: main — per block: 128 tokens, sweep all 8192 codes with WMMA,
// TDM double-buffered B tiles, running argmin, then gather + partial MSE.
// ---------------------------------------------------------------------------
__global__ __launch_bounds__(256) void vq_main(
    const float* __restrict__ x, const float* __restrict__ embed,
    const __bf16* __restrict__ eb_t, const float* __restrict__ enorm,
    float* __restrict__ out, float* __restrict__ partials) {
  __shared__ __align__(32) __bf16 btile[2][32 * DIM_D];  // 2 x 16KB
  __shared__ float etile[2][32];
  __shared__ unsigned long long best[128];
  __shared__ float rsum[256];

  const int t    = threadIdx.x;
  const int wave = t >> 5;
  const int lane = t & 31;
  const int half = lane >> 4;   // which 16-lane half
  const int l15  = lane & 15;
  const int tokBase = blockIdx.x * 128;
  const int waveTok = tokBase + wave * 16;

  if (t < 128) best[t] = ~0ULL;

  // ---- A fragments (16 tokens x 256 d), register-resident for whole sweep.
  // k order used here for A and B (contiguous): lanes 0-15 hold K {0..15}, lanes 16-31
  // hold {16..31} within each 32-wide K slice.
  const float* xrow = x + (size_t)(waveTok + l15) * DIM_D;
  v16bf a[8];
#pragma unroll
  for (int j = 0; j < 8; ++j) {
    const int d0 = j * 32;
    const float4* f1 = (const float4*)(xrow + d0 + 16 * half);
    const float4* f2 = (const float4*)(xrow + d0 + 8 + 16 * half);
    float4 u0 = f1[0], u1 = f1[1];
    float4 v0 = f2[0], v1 = f2[1];
    a[j][0]  = (__bf16)u0.x; a[j][1]  = (__bf16)u0.y;
    a[j][2]  = (__bf16)u0.z; a[j][3]  = (__bf16)u0.w;
    a[j][4]  = (__bf16)u1.x; a[j][5]  = (__bf16)u1.y;
    a[j][6]  = (__bf16)u1.z; a[j][7]  = (__bf16)u1.w;
    a[j][8]  = (__bf16)v0.x; a[j][9]  = (__bf16)v0.y;
    a[j][10] = (__bf16)v0.z; a[j][11] = (__bf16)v0.w;
    a[j][12] = (__bf16)v1.x; a[j][13] = (__bf16)v1.y;
    a[j][14] = (__bf16)v1.z; a[j][15] = (__bf16)v1.w;
  }

  const unsigned ldsOff0 = (unsigned)(uintptr_t)&btile[0][0];
  const unsigned ldsOff1 = (unsigned)(uintptr_t)&btile[1][0];

  // ---- prologue: TDM tile 0 into buf0
  if (wave == 0) tdm_load_16k(eb_t, ldsOff0);
  if (t < 32) etile[0][t] = enorm[t];
  if (wave == 0) __builtin_amdgcn_s_wait_tensorcnt(0);
  __syncthreads();

  float    minv[8];
  unsigned mini[8];
#pragma unroll
  for (int r = 0; r < 8; ++r) { minv[r] = 3.4e38f; mini[r] = 0u; }

  for (int it = 0; it < DIM_K / 32; ++it) {
    const int p = it & 1;
    // issue DMA for next tile into back buffer (safe: end-of-iteration
    // barrier of it-1 means nobody is still reading buf[p^1])
    float se = 0.f;
    if (it + 1 < DIM_K / 32) {
      if (wave == 0)
        tdm_load_16k(eb_t + (size_t)(it + 1) * (32 * DIM_D),
                     p ? ldsOff0 : ldsOff1);
      if (t < 32) se = enorm[(it + 1) * 32 + t];
    }

    // compute: 16 tokens x 32 codes = two 16x16 C tiles, K-chained over D=256
    v8f c0 = {0.f, 0.f, 0.f, 0.f, 0.f, 0.f, 0.f, 0.f};
    v8f c1 = {0.f, 0.f, 0.f, 0.f, 0.f, 0.f, 0.f, 0.f};
    const __bf16* b0p = &btile[p][(size_t)l15 * DIM_D + half * 16];
    const __bf16* b1p = b0p + 16 * DIM_D;
    v16bf b0 = *(const v16bf*)(b0p);
    v16bf b1 = *(const v16bf*)(b1p);
#pragma unroll
    for (int j = 0; j < 8; ++j) {
      v16bf nb0 = b0, nb1 = b1;
      if (j < 7) {  // prefetch next fragments before this step's WMMAs
        nb0 = *(const v16bf*)(b0p + (j + 1) * 32);
        nb1 = *(const v16bf*)(b1p + (j + 1) * 32);
      }
      c0 = __builtin_amdgcn_wmma_f32_16x16x32_bf16(false, a[j], false, b0,
                                                   (short)0, c0, false, false);
      c1 = __builtin_amdgcn_wmma_f32_16x16x32_bf16(false, a[j], false, b1,
                                                   (short)0, c1, false, false);
      b0 = nb0; b1 = nb1;
    }

    // dist = ||e||^2 - 2*score ; running argmin per (token, lane-column)
    const float e0 = etile[p][l15];
    const float e1 = etile[p][16 + l15];
    const unsigned i0 = (unsigned)(it * 32 + l15);
    const unsigned i1 = i0 + 16u;
#pragma unroll
    for (int r = 0; r < 8; ++r) {
      float v0 = fmaf(c0[r], -2.f, e0);
      if (v0 < minv[r]) { minv[r] = v0; mini[r] = i0; }
      float v1 = fmaf(c1[r], -2.f, e1);
      if (v1 < minv[r]) { minv[r] = v1; mini[r] = i1; }
    }

    if (it + 1 < DIM_K / 32 && t < 32) etile[p ^ 1][t] = se;
    if (wave == 0) __builtin_amdgcn_s_wait_tensorcnt(0);
    __syncthreads();
  }

  // ---- cross-lane argmin within each 16-lane half (C layout: VGPR r,
  // lanes 0-15 -> M=r, lanes 16-31 -> M=r+8; N = lane&15)
#pragma unroll
  for (int r = 0; r < 8; ++r) {
    float v = minv[r];
    unsigned i = mini[r];
#pragma unroll
    for (int m = 1; m <= 8; m <<= 1) {
      float ov = __shfl_xor(v, m, 32);
      unsigned oi = __shfl_xor(i, m, 32);
      if (ov < v || (ov == v && oi < i)) { v = ov; i = oi; }
    }
    if (l15 == 0) {
      unsigned kb = __float_as_uint(v);
      kb = (kb & 0x80000000u) ? ~kb : (kb | 0x80000000u);
      unsigned long long packed = ((unsigned long long)kb << 32) | i;
      atomicMin(&best[wave * 16 + r + 8 * half], packed);
    }
  }
  __syncthreads();

  // ---- finalize: indices, gather (exact f32 embed), partial MSE
  if (t < 128) {
    unsigned idx = (unsigned)(best[t] & 0xFFFFFFFFu);
    out[IND_OFF + tokBase + t] = (float)idx;
  }
  float local = 0.f;
  const int d = t;  // 0..255
  const float* erow = embed + (size_t)d * DIM_K;
  for (int i = 0; i < 128; ++i) {
    unsigned idx = (unsigned)(best[i] & 0xFFFFFFFFu);
    float q  = erow[idx];
    float xv = x[(size_t)(tokBase + i) * DIM_D + d];
    float dd = q - xv;
    local = fmaf(dd, dd, local);
    out[(size_t)(tokBase + i) * DIM_D + d] = q;
  }
  rsum[t] = local;
  __syncthreads();
  for (int s = 128; s > 0; s >>= 1) {
    if (t < s) rsum[t] += rsum[t + s];
    __syncthreads();
  }
  if (t == 0) partials[blockIdx.x] = rsum[0];
}

// ---------------------------------------------------------------------------
// Kernel 4: deterministic reduction of 128 block partials -> diff scalar
// ---------------------------------------------------------------------------
__global__ __launch_bounds__(128) void vq_diff(
    const float* __restrict__ partials, float* __restrict__ out) {
  __shared__ float s[128];
  const int t = threadIdx.x;
  s[t] = partials[t];
  __syncthreads();
  for (int k = 64; k > 0; k >>= 1) {
    if (t < k) s[t] += s[t + k];
    __syncthreads();
  }
  if (t == 0) out[DIFF_OFF] = s[0] * (1.f / 4194304.f);
}

// ---------------------------------------------------------------------------
extern "C" void kernel_launch(void* const* d_in, const int* in_sizes, int n_in,
                              void* d_out, int out_size, void* d_ws,
                              size_t ws_size, hipStream_t stream) {
  (void)in_sizes; (void)n_in; (void)out_size; (void)ws_size;
  const float* x     = (const float*)d_in[0];  // [16,32,32,256]
  const float* embed = (const float*)d_in[1];  // [256,8192]
  float* out = (float*)d_out;

  char* ws = (char*)d_ws;
  __bf16* eb_t  = (__bf16*)ws;                               // 4 MB
  float*  enorm = (float*)(ws + (size_t)DIM_K * DIM_D * 2);  // 32 KB
  float*  parts =
      (float*)(ws + (size_t)DIM_K * DIM_D * 2 + (size_t)DIM_K * 4);

  vq_transpose_convert<<<dim3(DIM_K / 32, DIM_D / 32), dim3(32, 8), 0,
                         stream>>>(embed, eb_t);
  vq_enorm<<<DIM_K / 256, 256, 0, stream>>>(embed, enorm);
  vq_main<<<N_TOK / 128, 256, 0, stream>>>(x, embed, eb_t, enorm, out, parts);
  vq_diff<<<1, 128, 0, stream>>>(parts, out);
}
